// MultiHeadAttention_68006512165003
// MI455X (gfx1250) — matrix operand order repaired and checked by reading, not compiled, not run on hardware
//
#include <hip/hip_runtime.h>
#include <hip/hip_bf16.h>

// ---------------- problem constants ----------------
constexpr int B_SZ    = 4;
constexpr int S_LEN   = 2048;
constexpr int D_MODEL = 1024;
constexpr int H_CNT   = 16;
constexpr int DH      = 64;                  // head dim
constexpr int M_ROWS  = B_SZ * S_LEN;        // 8192

// ---------------- CDNA5 WMMA types ----------------
typedef __attribute__((ext_vector_type(16))) __bf16 v16bf;
typedef __attribute__((ext_vector_type(8)))  float  v8f;

union FragU { uint4 u[2]; v16bf v; };

__device__ inline v8f wmma_bf16(v16bf a, v16bf b, v8f c) {
    // D = A(16x32 bf16) x B(32x16 bf16) + C(16x16 f32)
    return __builtin_amdgcn_wmma_f32_16x16x32_bf16(
        /*neg_a=*/false, a, /*neg_b=*/false, b,
        /*c_mod=*/(short)0, c, /*reuse_a=*/false, /*reuse_b=*/false);
}

__device__ inline unsigned short f2bf(float x) {
    unsigned int u = __float_as_uint(x);
    u += 0x7FFFu + ((u >> 16) & 1u);         // round-to-nearest-even
    return (unsigned short)(u >> 16);
}

// ---- CDNA5 async memory->LDS copy (GLOBAL_LOAD_ASYNC_TO_LDS_B128, ASYNCcnt) ----
// vdst VGPR = 32-bit LDS byte offset (low 32 bits of a generic __shared__ pointer).
__device__ inline void async_ld_lds_b128(void* lds_dst, const void* gsrc) {
    unsigned loff = (unsigned)(unsigned long long)lds_dst;
    asm volatile("global_load_async_to_lds_b128 %0, %1, off"
                 :: "v"(loff), "v"(gsrc) : "memory");
}
// Async loads complete IN ORDER -> waiting for <=N leaves the newest N in flight
// (double-buffer pipelining: next tile stays outstanding behind the WMMA work).
template<int N> __device__ inline void wait_async() {
    asm volatile("s_wait_asynccnt %0" :: "i"(N) : "memory");
}

// A-matrix fragment (16x32, bf16) from LDS laid out [m][k] with row `stride`.
// ISA 7.12.2: lane m = L%16; lanes<16 hold K = {kb..kb+7, 16+kb..16+kb+7}, kb = 8*(L>=16).
__device__ inline v16bf load_a_frag(const unsigned short* base, int stride, int k0, int lane) {
    const int m  = lane & 15;
    const int kb = (lane >> 4) ? 8 : 0;
    FragU f;
    f.u[0] = *(const uint4*)(base + m * stride + k0 + kb);        // dwords 0..3
    f.u[1] = *(const uint4*)(base + m * stride + k0 + kb + 16);   // dwords 4..7
    return f.v;
}

// B-matrix fragment (32x16, bf16) from LDS laid out transposed [n][k] with row `stride`.
// lane n = L%16; K = {kh..kh+7, 16+kh..16+kh+7}, kh = 8*(L>=16) (matches A-frag order).
__device__ inline v16bf load_b_frag(const unsigned short* base, int stride, int k0, int lane) {
    const int n  = lane & 15;
    const int kh = (lane >> 4) ? 8 : 0;
    FragU f;
    f.u[0] = *(const uint4*)(base + n * stride + k0 + kh);
    f.u[1] = *(const uint4*)(base + n * stride + k0 + kh + 16);
    return f.v;
}

// ---------------- casts fp32 -> bf16 ----------------
__global__ void mha_cast_bf16(const float* __restrict__ src,
                              unsigned short* __restrict__ dst, int n) {
    int i = blockIdx.x * blockDim.x + threadIdx.x;
    if (i < n) dst[i] = f2bf(src[i]);
}

// W[K,N] -> Wt[N,K] bf16 (one-time; makes GEMM B-tiles straight b128 copies)
__global__ void mha_cast_tr_bf16(const float* __restrict__ W,
                                 unsigned short* __restrict__ Wt) {
    int i = blockIdx.x * blockDim.x + threadIdx.x;
    if (i < D_MODEL * D_MODEL) {
        int k = i >> 10, n = i & (D_MODEL - 1);       // coalesced read of W
        Wt[(size_t)n * D_MODEL + k] = f2bf(W[i]);
    }
}

// ---------------- tiled bf16 WMMA GEMM: C = A[8192,1024] @ Bt[1024,1024]^T ----
// 128x128 block tile, 256 threads = 8 waves (4m x 2n); wave = 32 rows x 64 cols:
// 2 A-frags + 4 B-frags -> 8 WMMAs per k-step. Double-buffered async->LDS pipeline.
// MODE 0: bf16 row-major; 1: f32 + bias; 2: bf16 transposed per head
//   ([B,S,H*Dh] -> [B,H,Dh,S], used for V so flash PV B-tiles are contiguous).
template<int MODE>
__global__ __launch_bounds__(256) void mha_gemm_bf16(
    const unsigned short* __restrict__ A,  const unsigned short* __restrict__ Bt,
    unsigned short* __restrict__ Cb, float* __restrict__ Cf,
    const float* __restrict__ bias)
{
    constexpr int N = D_MODEL, K = D_MODEL;
    __shared__ alignas(16) unsigned short As[2][128 * 32];   // [m][k]
    __shared__ alignas(16) unsigned short Bs[2][128 * 32];   // [n][k]

    const int tid  = threadIdx.x;
    const int lane = tid & 31;
    const int wave = tid >> 5;
    const int wm   = wave >> 1, wn = wave & 1;
    const int m0   = blockIdx.y * 128;
    const int n0   = blockIdx.x * 128;

    auto issue_tile = [&](int buf, int k0) {   // 4 async b128 loads per thread
        #pragma unroll
        for (int j = 0; j < 2; ++j) {
            const int i = tid + j * 256;
            const int r = i >> 2, c = (i & 3) * 8;
            async_ld_lds_b128(As[buf] + i * 8, A  + (size_t)(m0 + r) * K + k0 + c);
            async_ld_lds_b128(Bs[buf] + i * 8, Bt + (size_t)(n0 + r) * K + k0 + c);
        }
    };

    v8f acc[2][4] = {};
    issue_tile(0, 0);
    int buf = 0;
    for (int k0 = 0; k0 < K; k0 += 32) {
        const bool more = (k0 + 32 < K);
        if (more) { issue_tile(buf ^ 1, k0 + 32); wait_async<4>(); }
        else      { wait_async<0>(); }
        __syncthreads();                         // current tile visible to all waves

        const v16bf a0 = load_a_frag(As[buf] + (wm * 32) * 32,      32, 0, lane);
        const v16bf a1 = load_a_frag(As[buf] + (wm * 32 + 16) * 32, 32, 0, lane);
        #pragma unroll
        for (int t = 0; t < 4; ++t) {
            const v16bf b = load_b_frag(Bs[buf] + (wn * 64 + t * 16) * 32, 32, 0, lane);
            acc[0][t] = wmma_bf16(a0, b, acc[0][t]);
            acc[1][t] = wmma_bf16(a1, b, acc[1][t]);
        }
        __syncthreads();                         // reads done before next overwrite
        buf ^= 1;
    }

    const int ln = lane & 15, lh = lane >> 4;
    #pragma unroll
    for (int u = 0; u < 2; ++u) {
        #pragma unroll
        for (int t = 0; t < 4; ++t) {
            const int col = n0 + wn * 64 + t * 16 + ln;
            #pragma unroll
            for (int r = 0; r < 8; ++r) {
                const int row = m0 + wm * 32 + u * 16 + r + 8 * lh;  // ISA C-layout
                const float val = acc[u][t][r];
                if (MODE == 1) {
                    Cf[(size_t)row * N + col] = val + bias[col];
                } else if (MODE == 0) {
                    Cb[(size_t)row * N + col] = f2bf(val);
                } else {
                    const int bb = row >> 11, ss = row & (S_LEN - 1);
                    const int hh = col >> 6,  dd = col & (DH - 1);
                    Cb[(((size_t)bb * H_CNT + hh) * DH + dd) * S_LEN + ss] = f2bf(val);
                }
            }
        }
    }
}

// ---------------- causal flash attention, bf16 WMMA, f32 softmax ------------
// grid: (S/64, H, B); block: 128 threads (4 waves, wave = 16 query rows).
// Vt is [B][H][Dh][S] so PV B-tiles are contiguous. K/V tiles double-buffered
// through the async engine; Q tile rides in front of KV0 in the same queue.
__global__ __launch_bounds__(128) void mha_flash(
    const unsigned short* __restrict__ Q, const unsigned short* __restrict__ Kg,
    const unsigned short* __restrict__ Vt, unsigned short* __restrict__ O)
{
    __shared__ alignas(16) unsigned short Qs[64 * 64];        // [qrow][d]
    __shared__ alignas(16) unsigned short Ks[2][64 * 64];     // [key][d]  (QK^T B-frags)
    __shared__ alignas(16) unsigned short Vs[2][64 * 64];     // [d][key]  (PV  B-frags)
    __shared__ alignas(16) unsigned short Ps[4][16 * 64];     // per-wave P [qrow][key]

    const int tid  = threadIdx.x;
    const int lane = tid & 31;
    const int wave = tid >> 5;
    const int ln = lane & 15, lh = lane >> 4;
    const int q0 = blockIdx.x * 64;
    const int h  = blockIdx.y;
    const int b  = blockIdx.z;
    const size_t rowBase = (size_t)b * S_LEN * D_MODEL;
    const size_t vtBase  = ((size_t)b * H_CNT + h) * DH * S_LEN;
    const int hOff = h * DH;

    auto issue_kv = [&](int buf, int kt0) {    // 8 async b128 loads per thread
        #pragma unroll
        for (int j = 0; j < 4; ++j) {
            const int i = tid + j * 128;
            const int r = i >> 3, c = (i & 7) * 8;
            async_ld_lds_b128(Ks[buf] + i * 8, Kg + rowBase + (size_t)(kt0 + r) * D_MODEL + hOff + c);
            async_ld_lds_b128(Vs[buf] + i * 8, Vt + vtBase + (size_t)r * S_LEN + kt0 + c);
        }
    };

    // Q tile (4 loads/thread), then KV tile 0 queued right behind it
    #pragma unroll
    for (int j = 0; j < 4; ++j) {
        const int i = tid + j * 128;
        const int r = i >> 3, c = (i & 7) * 8;
        async_ld_lds_b128(Qs + i * 8, Q + rowBase + (size_t)(q0 + r) * D_MODEL + hOff + c);
    }
    issue_kv(0, 0);
    wait_async<8>();                             // in-order: Q complete, KV0 in flight
    __syncthreads();
    const v16bf qf0 = load_a_frag(Qs + wave * 16 * 64, 64, 0,  lane);
    const v16bf qf1 = load_a_frag(Qs + wave * 16 * 64, 64, 32, lane);

    v8f oacc[4] = {};
    float m_run[8], l_run[8];
    #pragma unroll
    for (int r = 0; r < 8; ++r) { m_run[r] = -INFINITY; l_run[r] = 0.0f; }

    int buf = 0;
    for (int kt0 = 0; kt0 <= q0; kt0 += 64) {
        const bool more = (kt0 + 64 <= q0);
        if (more) { issue_kv(buf ^ 1, kt0 + 64); wait_async<8>(); }
        else      { wait_async<0>(); }
        __syncthreads();                         // KV[buf] visible to all waves

        // S = Q @ K^T  (Dh = 64 -> two K=32 WMMA steps per 16x16 key subtile)
        v8f s[4];
        #pragma unroll
        for (int t = 0; t < 4; ++t) {
            v8f z = {};
            z = wmma_bf16(qf0, load_b_frag(Ks[buf] + t * 16 * 64, 64, 0,  lane), z);
            z = wmma_bf16(qf1, load_b_frag(Ks[buf] + t * 16 * 64, 64, 32, lane), z);
            s[t] = z;
        }

        // scale + causal mask (diagonal tile only)
        const bool diag = (kt0 == q0);
        #pragma unroll
        for (int t = 0; t < 4; ++t) {
            #pragma unroll
            for (int r = 0; r < 8; ++r) {
                float v = s[t][r] * 0.125f;                 // 1/sqrt(64)
                if (diag) {
                    const int rowl = wave * 16 + r + 8 * lh;
                    const int coll = t * 16 + ln;
                    if (coll > rowl) v = -INFINITY;
                }
                s[t][r] = v;
            }
        }

        // online softmax: row max / sum across the 16-lane half sharing a row
        float mx[8];
        #pragma unroll
        for (int r = 0; r < 8; ++r) {
            float m = fmaxf(fmaxf(s[0][r], s[1][r]), fmaxf(s[2][r], s[3][r]));
            m = fmaxf(m, __shfl_xor(m, 1, 32));
            m = fmaxf(m, __shfl_xor(m, 2, 32));
            m = fmaxf(m, __shfl_xor(m, 4, 32));
            m = fmaxf(m, __shfl_xor(m, 8, 32));
            mx[r] = m;
        }
        float corr[8];
        #pragma unroll
        for (int r = 0; r < 8; ++r) {
            const float mn = fmaxf(m_run[r], mx[r]);
            corr[r] = __expf(m_run[r] - mn);
            m_run[r] = mn;
        }
        #pragma unroll
        for (int t = 0; t < 4; ++t)
            #pragma unroll
            for (int r = 0; r < 8; ++r)
                s[t][r] = __expf(s[t][r] - m_run[r]);
        #pragma unroll
        for (int r = 0; r < 8; ++r) {
            float sm = s[0][r] + s[1][r] + s[2][r] + s[3][r];
            sm += __shfl_xor(sm, 1, 32);
            sm += __shfl_xor(sm, 2, 32);
            sm += __shfl_xor(sm, 4, 32);
            sm += __shfl_xor(sm, 8, 32);
            l_run[r] = l_run[r] * corr[r] + sm;
            #pragma unroll
            for (int t = 0; t < 4; ++t) oacc[t][r] *= corr[r];
        }

        // P: C-layout -> LDS -> A-fragment layout
        #pragma unroll
        for (int t = 0; t < 4; ++t)
            #pragma unroll
            for (int r = 0; r < 8; ++r)
                Ps[wave][(r + 8 * lh) * 64 + t * 16 + ln] = f2bf(s[t][r]);
        __syncthreads();

        const v16bf p0 = load_a_frag(Ps[wave], 64, 0,  lane);
        const v16bf p1 = load_a_frag(Ps[wave], 64, 32, lane);
        #pragma unroll
        for (int t = 0; t < 4; ++t) {
            oacc[t] = wmma_bf16(p0, load_b_frag(Vs[buf] + t * 16 * 64, 64, 0,  lane), oacc[t]);
            oacc[t] = wmma_bf16(p1, load_b_frag(Vs[buf] + t * 16 * 64, 64, 32, lane), oacc[t]);
        }
        __syncthreads();                         // reads done before next overwrite
        buf ^= 1;
    }

    // O = O / l, stored bf16 in [B, S, H*Dh] so the output GEMM reads it directly
    float inv_l[8];
    #pragma unroll
    for (int r = 0; r < 8; ++r) inv_l[r] = 1.0f / l_run[r];
    #pragma unroll
    for (int t = 0; t < 4; ++t)
        #pragma unroll
        for (int r = 0; r < 8; ++r) {
            const int row = q0 + wave * 16 + r + 8 * lh;
            const int col = hOff + t * 16 + ln;
            O[rowBase + (size_t)row * D_MODEL + col] = f2bf(oacc[t][r] * inv_l[r]);
        }
}

// ---------------- host launcher ----------------
extern "C" void kernel_launch(void* const* d_in, const int* in_sizes, int n_in,
                              void* d_out, int out_size, void* d_ws, size_t ws_size,
                              hipStream_t stream) {
    const float* x  = (const float*)d_in[0];
    const float* Wq = (const float*)d_in[1];
    const float* Wk = (const float*)d_in[2];
    const float* Wv = (const float*)d_in[3];
    const float* Wo = (const float*)d_in[4];
    const float* bo = (const float*)d_in[5];
    float* out = (float*)d_out;
    char* ws = (char*)d_ws;

    constexpr size_t SZ_X = (size_t)M_ROWS * D_MODEL * sizeof(unsigned short);  // 16 MiB
    constexpr size_t SZ_W = (size_t)D_MODEL * D_MODEL * sizeof(unsigned short); //  2 MiB
    unsigned short* xb  = (unsigned short*)(ws);
    unsigned short* wqb = (unsigned short*)(ws + SZ_X);                // all [N,K] transposed
    unsigned short* wkb = (unsigned short*)(ws + SZ_X + 1 * SZ_W);
    unsigned short* wvb = (unsigned short*)(ws + SZ_X + 2 * SZ_W);
    unsigned short* wob = (unsigned short*)(ws + SZ_X + 3 * SZ_W);
    unsigned short* qb  = (unsigned short*)(ws + SZ_X + 4 * SZ_W);     // [B,S,D]
    unsigned short* kb  = (unsigned short*)(ws + 2 * SZ_X + 4 * SZ_W); // [B,S,D]
    unsigned short* vtb = (unsigned short*)(ws + 3 * SZ_X + 4 * SZ_W); // [B,H,Dh,S]
    unsigned short* aob = (unsigned short*)(ws + 4 * SZ_X + 4 * SZ_W); // [B,S,D]

    const int nx = M_ROWS * D_MODEL;      // 8388608
    const int nw = D_MODEL * D_MODEL;     // 1048576
    mha_cast_bf16<<<(nx + 255) / 256, 256, 0, stream>>>(x, xb, nx);
    mha_cast_tr_bf16<<<(nw + 255) / 256, 256, 0, stream>>>(Wq, wqb);
    mha_cast_tr_bf16<<<(nw + 255) / 256, 256, 0, stream>>>(Wk, wkb);
    mha_cast_tr_bf16<<<(nw + 255) / 256, 256, 0, stream>>>(Wv, wvb);
    mha_cast_tr_bf16<<<(nw + 255) / 256, 256, 0, stream>>>(Wo, wob);

    dim3 gg(D_MODEL / 128, M_ROWS / 128);  // (8, 64)
    mha_gemm_bf16<0><<<gg, 256, 0, stream>>>(xb, wqb, qb,  nullptr, nullptr);
    mha_gemm_bf16<0><<<gg, 256, 0, stream>>>(xb, wkb, kb,  nullptr, nullptr);
    mha_gemm_bf16<2><<<gg, 256, 0, stream>>>(xb, wvb, vtb, nullptr, nullptr);

    dim3 fg(S_LEN / 64, H_CNT, B_SZ);      // (32, 16, 4)
    mha_flash<<<fg, 128, 0, stream>>>(qb, kb, vtb, aob);

    mha_gemm_bf16<1><<<gg, 256, 0, stream>>>(aob, wob, nullptr, out, bo);
}
